// FF_42571715838428
// MI455X (gfx1250) — compile-verified, matrix operand order checked by reading, not run on hardware
//
#include <hip/hip_runtime.h>
#include <hip/hip_bf16.h>
#include <math.h>

// Problem sizes (fixed by the reference)
#define TOKENS 16384   // B*S = 32*512
#define DIN    768
#define DHID   3072
#define DOUT   768

#define LDSP   40      // padded LDS row (32 data + 8 pad bf16) -> 80B rows, bank-spread

typedef __attribute__((ext_vector_type(16))) __bf16 bf16x16;
typedef __attribute__((ext_vector_type(8)))  __bf16 bf16x8;
typedef __attribute__((ext_vector_type(8)))  float  v8f;

#ifndef __has_builtin
#define __has_builtin(x) 0
#endif
#if __has_builtin(__builtin_amdgcn_global_load_async_to_lds_b128) && \
    __has_builtin(__builtin_amdgcn_s_wait_asynccnt)
#define HAS_ASYNC_LDS 1
#else
#define HAS_ASYNC_LDS 0
#endif

#if HAS_ASYNC_LDS
// Builtin signature (from hipcc diagnostic): param 0 is a non-const pointer to
// 'int __attribute__((vector_size(16)))' in the global address space.
typedef int v4i_vs __attribute__((__vector_size__(16)));
typedef __attribute__((address_space(1))) v4i_vs* as1_v4i;
typedef __attribute__((address_space(3))) v4i_vs* as3_v4i;
#endif

// ---------- helpers ----------
__device__ __forceinline__ unsigned short f32_to_bf16_rne(float f) {
    unsigned u = __float_as_uint(f);
    u += 0x7FFFu + ((u >> 16) & 1u);       // round-to-nearest-even
    return (unsigned short)(u >> 16);
}
__device__ __forceinline__ unsigned pack2(float lo, float hi) {
    return (unsigned)f32_to_bf16_rne(lo) | ((unsigned)f32_to_bf16_rne(hi) << 16);
}

// copy 16 bytes global -> LDS (async on gfx1250 if the builtin is available)
__device__ __forceinline__ void cp16_g2l(const unsigned short* g, unsigned short* l) {
#if HAS_ASYNC_LDS
    __builtin_amdgcn_global_load_async_to_lds_b128(
        (as1_v4i)(void*)g,   // C-style cast drops const, addrspacecast to AS1
        (as3_v4i)(void*)l,   // addrspacecast generic -> LDS
        0, 0);
#else
    *(uint4*)l = *(const uint4*)g;
#endif
}
__device__ __forceinline__ void wait_async_lds() {
#if HAS_ASYNC_LDS
    __builtin_amdgcn_s_wait_asynccnt(0);
#endif
}

// Load a 16x32 (MxK) bf16 WMMA fragment from an LDS tile stored [row][K(0..31)] w/ LDSP pad.
// Per ISA 7.12.2: lane l holds row (l&15); lanes 0-15 take K [0..7]+[16..23],
// lanes 16-31 take K [8..15]+[24..31]  -> two contiguous 16B LDS reads per lane.
__device__ __forceinline__ bf16x16 load_frag(const unsigned short* s, int rowBase, int lane) {
    const int r  = rowBase + (lane & 15);
    const int c0 = (lane >> 4) << 3;       // 0 or 8
    const bf16x8* p = (const bf16x8*)(s + r * LDSP + c0);
    bf16x8 lo = p[0];                      // K = c0 .. c0+7
    bf16x8 hi = p[2];                      // K = c0+16 .. c0+23
    return __builtin_shufflevector(lo, hi, 0,1,2,3,4,5,6,7,8,9,10,11,12,13,14,15);
}

__device__ __forceinline__ void compute_step(const unsigned short* a, const unsigned short* b,
                                             int wm, int wn, int lane, v8f acc[4][2]) {
    bf16x16 af[4];
    bf16x16 bfr[2];
#pragma unroll
    for (int mi = 0; mi < 4; ++mi) af[mi]  = load_frag(a, wm * 64 + mi * 16, lane);
#pragma unroll
    for (int ni = 0; ni < 2; ++ni) bfr[ni] = load_frag(b, wn * 32 + ni * 16, lane);
#pragma unroll
    for (int mi = 0; mi < 4; ++mi)
#pragma unroll
        for (int ni = 0; ni < 2; ++ni)
            acc[mi][ni] = __builtin_amdgcn_wmma_f32_16x16x32_bf16(
                false, af[mi], false, bfr[ni], (short)0, acc[mi][ni], false, false);
}

// ---------- prep 0: elementwise f32 -> bf16 (vectorized, 8 elems/thread) ----------
__global__ __launch_bounds__(256)
void cvt_f32_to_bf16(const float* __restrict__ src, unsigned short* __restrict__ dst, int n8) {
    const int i = blockIdx.x * 256 + threadIdx.x;
    if (i >= n8) return;
    const float4* s4 = (const float4*)src + (size_t)i * 2;
    float4 a = s4[0], b = s4[1];
    uint4 u = { pack2(a.x, a.y), pack2(a.z, a.w), pack2(b.x, b.y), pack2(b.z, b.w) };
    ((uint4*)dst)[i] = u;
}

// ---------- prep 1: W_out [DHID x DOUT] f32 -> W_outT [DOUT x DHID] bf16 ----------
__global__ __launch_bounds__(256)
void wout_transpose_bf16(const float* __restrict__ W, unsigned short* __restrict__ WT) {
    __shared__ float tile[32][33];
    const int o0 = blockIdx.x * 32;
    const int h0 = blockIdx.y * 32;
    const int tx = threadIdx.x & 31;
    const int ty = threadIdx.x >> 5;       // 0..7
#pragma unroll
    for (int i = 0; i < 32; i += 8)
        tile[ty + i][tx] = W[(size_t)(h0 + ty + i) * DOUT + o0 + tx];
    __syncthreads();
#pragma unroll
    for (int i = 0; i < 32; i += 8)
        WT[(size_t)(o0 + ty + i) * DHID + h0 + tx] = f32_to_bf16_rne(tile[tx][ty + i]);
}

// ---------- unified bf16 GEMM: C[128x128] tile, K-loop in 32-steps ----------
// A: [M x KDIM] bf16 row-major, B: [N x KDIM] bf16 row-major (i.e. B^T layout).
// DO_GELU: epilogue adds bias + exact GELU, stores bf16 to OutBf (stride OSTRIDE);
// else stores f32 to OutF (stride OSTRIDE).
template <int KDIM, bool DO_GELU, int OSTRIDE>
__global__ __launch_bounds__(256)
void gemm_bf16_wmma(const unsigned short* __restrict__ A,
                    const unsigned short* __restrict__ B,
                    const float* __restrict__ bias,
                    unsigned short* __restrict__ OutBf,
                    float* __restrict__ OutF) {
    __shared__ unsigned short sA[2][128 * LDSP];
    __shared__ unsigned short sB[2][128 * LDSP];

    const int tid   = threadIdx.x;
    const int lane  = tid & 31;
    const int wid   = tid >> 5;
    const int wm    = wid >> 2;            // 0..1  (64 rows each)
    const int wn    = wid & 3;             // 0..3  (32 cols each)
    const int tBase = blockIdx.x * 128;
    const int nBase = blockIdx.y * 128;
    const int srow  = tid >> 1;            // staging row 0..127
    const int sel   = (tid & 1) << 4;      // staging element offset: 0 or 16

    v8f acc[4][2];
#pragma unroll
    for (int mi = 0; mi < 4; ++mi)
#pragma unroll
        for (int ni = 0; ni < 2; ++ni) acc[mi][ni] = (v8f){0,0,0,0,0,0,0,0};

    const unsigned short* ga = A + (size_t)(tBase + srow) * KDIM + sel;
    const unsigned short* gb = B + (size_t)(nBase + srow) * KDIM + sel;
    const int ldsOff = srow * LDSP + sel;

    // stage tile 0
    cp16_g2l(ga,     &sA[0][ldsOff]);
    cp16_g2l(ga + 8, &sA[0][ldsOff + 8]);
    cp16_g2l(gb,     &sB[0][ldsOff]);
    cp16_g2l(gb + 8, &sB[0][ldsOff + 8]);
    wait_async_lds();
    __syncthreads();

    const int KT = KDIM / 32;
    for (int kt = 0; kt < KT; ++kt) {
        const int buf = kt & 1;
        if (kt + 1 < KT) {                 // async-copy next tile into other buffer
            const int k1 = (kt + 1) * 32;
            cp16_g2l(ga + k1,     &sA[buf ^ 1][ldsOff]);
            cp16_g2l(ga + k1 + 8, &sA[buf ^ 1][ldsOff + 8]);
            cp16_g2l(gb + k1,     &sB[buf ^ 1][ldsOff]);
            cp16_g2l(gb + k1 + 8, &sB[buf ^ 1][ldsOff + 8]);
        }
        compute_step(sA[buf], sB[buf], wm, wn, lane, acc);
        if (kt + 1 < KT) wait_async_lds();
        __syncthreads();
    }

    // epilogue
#pragma unroll
    for (int ni = 0; ni < 2; ++ni) {
        const int n = nBase + wn * 32 + ni * 16 + (lane & 15);
        float bv = 0.0f;
        if (DO_GELU) bv = bias[n];
#pragma unroll
        for (int mi = 0; mi < 4; ++mi) {
            const int trow = tBase + wm * 64 + mi * 16 + ((lane >> 4) << 3);
#pragma unroll
            for (int r = 0; r < 8; ++r) {
                if (DO_GELU) {
                    float v = acc[mi][ni][r] + bv;
                    v = 0.5f * v * (1.0f + erff(v * 0.70710678118654752f));
                    OutBf[(size_t)(trow + r) * OSTRIDE + n] = f32_to_bf16_rne(v);
                } else {
                    OutF[(size_t)(trow + r) * OSTRIDE + n] = acc[mi][ni][r];
                }
            }
        }
    }
}

// ---------- launcher ----------
extern "C" void kernel_launch(void* const* d_in, const int* in_sizes, int n_in,
                              void* d_out, int out_size, void* d_ws, size_t ws_size,
                              hipStream_t stream) {
    const float* x     = (const float*)d_in[0];   // [32,512,768]
    const float* w_in  = (const float*)d_in[1];   // [3072,768]
    const float* b_in  = (const float*)d_in[2];   // [3072]
    const float* w_out = (const float*)d_in[3];   // [3072,768]
    float* out = (float*)d_out;                   // [32,512,768]

    // workspace layout (bf16 elements)
    unsigned short* Hbuf = (unsigned short*)d_ws;                 // TOKENS*DHID
    unsigned short* WT   = Hbuf + (size_t)TOKENS * DHID;          // DOUT*DHID
    unsigned short* Xb   = WT   + (size_t)DOUT  * DHID;           // TOKENS*DIN
    unsigned short* Wb   = Xb   + (size_t)TOKENS * DIN;           // DHID*DIN

    // one-time conversions (bandwidth-trivial vs. the 155 GFLOP of GEMM)
    {
        const int n8x = (TOKENS * DIN) / 8;   // 1,572,864
        const int n8w = (DHID * DIN) / 8;     //   294,912
        cvt_f32_to_bf16<<<(n8x + 255) / 256, 256, 0, stream>>>(x, Xb, n8x);
        cvt_f32_to_bf16<<<(n8w + 255) / 256, 256, 0, stream>>>(w_in, Wb, n8w);
        wout_transpose_bf16<<<dim3(DOUT / 32, DHID / 32), 256, 0, stream>>>(w_out, WT);
    }

    // GEMM1: H = GELU(X @ W_in^T + b)   [TOKENS x DHID], K = DIN
    gemm_bf16_wmma<DIN, true, DHID>
        <<<dim3(TOKENS / 128, DHID / 128), 256, 0, stream>>>(Xb, Wb, b_in, Hbuf, nullptr);

    // GEMM2: O = H @ W_out              [TOKENS x DOUT], K = DHID
    gemm_bf16_wmma<DHID, false, DOUT>
        <<<dim3(TOKENS / 128, DOUT / 128), 256, 0, stream>>>(Hbuf, WT, nullptr, nullptr, out);
}
